// GeneralMACE_49520972923317
// MI455X (gfx1250) — matrix operand order repaired and checked by reading, not compiled, not run on hardware
//
#include <hip/hip_runtime.h>
#include <hip/hip_bf16.h>
#include <math.h>

// ---------------- problem constants ----------------
constexpr int NN   = 10000;     // nodes
constexpr int EE   = 160000;    // edges
constexpr int FF   = 128;       // channels
constexpr int NSPEC = 10;
constexpr int NLAY = 2;
constexpr int NB   = 8;
constexpr int HR   = 64;
constexpr int SHD  = 16;
constexpr float EPS_C = 0.5f;
constexpr float RMAX  = 5.0f;
constexpr float PI_F  = 3.14159265358979323846f;

typedef __attribute__((ext_vector_type(16))) __bf16 v16bf;
typedef __attribute__((ext_vector_type(8)))  __bf16 v8bf;
typedef __attribute__((ext_vector_type(8)))  float  v8f;

__constant__ int c_lmap[16] = {0,1,1,1,2,2,2,2,2,3,3,3,3,3,3,3};
__constant__ int c_off[4]   = {0,1,4,9};
__constant__ int c_ml[4]    = {1,3,5,7};

__device__ __forceinline__ float silu_f(float x) {
    return x / (1.0f + __expf(-x));
}

// 16-bit A-fragment from two contiguous 8-element halves (lo = elements 0..7, hi = 8..15).
__device__ __forceinline__ v16bf make_frag(const __bf16* p_lo, const __bf16* p_hi) {
    v8bf lo = *(const v8bf*)p_lo;
    v8bf hi = *(const v8bf*)p_hi;
    return __builtin_shufflevector(lo, hi, 0,1,2,3,4,5,6,7,8,9,10,11,12,13,14,15);
}

// ---------------- utility kernels ----------------
__global__ void k_zero_f32(float* __restrict__ p, long n) {
    long stride = (long)gridDim.x * blockDim.x;
    for (long i = (long)blockIdx.x * blockDim.x + threadIdx.x; i < n; i += stride)
        p[i] = 0.0f;
}

// W2 (NL,64,512) f32 -> transposed bf16 (NL,512,64): d[i][n][k] = s[i][k][n]
__global__ void k_cvt_w2t(const float* __restrict__ s, __bf16* __restrict__ d) {
    long n_tot = (long)NLAY * 512 * HR;
    long stride = (long)gridDim.x * blockDim.x;
    for (long i = (long)blockIdx.x * blockDim.x + threadIdx.x; i < n_tot; i += stride) {
        long lay = i / (512 * HR);
        long rem = i - lay * 512 * HR;
        long n = rem >> 6, k = rem & 63;
        d[i] = (__bf16)s[lay * HR * 512 + k * 512 + n];
    }
}

// lin/sc W (NL,NSPEC,4,F,F) f32 -> transposed bf16 blocks: d[b][g][f] = s[b][f][g]
__global__ void k_cvt_lint(const float* __restrict__ s, __bf16* __restrict__ d) {
    long n_tot = (long)NLAY * NSPEC * 4 * FF * FF;
    long stride = (long)gridDim.x * blockDim.x;
    for (long i = (long)blockIdx.x * blockDim.x + threadIdx.x; i < n_tot; i += stride) {
        long b = i / (FF * FF);
        long rem = i - b * FF * FF;
        long g = rem >> 7, f = rem & 127;
        d[i] = (__bf16)s[b * FF * FF + f * FF + g];
    }
}

// ---------------- edge preparation: SH(16) + Bessel radial(8) ----------------
__global__ __launch_bounds__(256)
void k_edge_prep(const float* __restrict__ vec, float* __restrict__ sh, float* __restrict__ rb) {
    int e = blockIdx.x * blockDim.x + threadIdx.x;
    if (e >= EE) return;
    float vx = vec[3*e+0], vy = vec[3*e+1], vz = vec[3*e+2];
    float r = sqrtf(vx*vx + vy*vy + vz*vz);
    r = fmaxf(r, 1e-9f);
    float inv = 1.0f / r;
    float x = vx*inv, y = vy*inv, z = vz*inv;

    const float s3 = 1.7320508075688772f, s5 = 2.23606797749979f, s7 = 2.6457513110645907f;
    const float s15 = 3.872983346207417f, s42 = 6.48074069840786f;
    const float s70 = 8.366600265340756f, s105 = 10.246950765959598f;

    float o[SHD];
    o[0]  = 1.0f;
    o[1]  = s3 * x;  o[2] = s3 * y;  o[3] = s3 * z;
    o[4]  = s15 * x * y;
    o[5]  = s15 * y * z;
    o[6]  = 0.5f * s5 * (3.0f*z*z - 1.0f);
    o[7]  = s15 * x * z;
    o[8]  = 0.5f * s15 * (x*x - y*y);
    o[9]  = 0.25f * s70 * y * (3.0f*x*x - y*y);
    o[10] = s105 * x * y * z;
    o[11] = 0.25f * s42 * y * (5.0f*z*z - 1.0f);
    o[12] = 0.5f * s7 * (5.0f*z*z*z - 3.0f*z);
    o[13] = 0.25f * s42 * x * (5.0f*z*z - 1.0f);
    o[14] = 0.5f * s105 * z * (x*x - y*y);
    o[15] = 0.25f * s70 * x * (x*x - 3.0f*y*y);
    float* op = sh + (size_t)e * SHD;
    #pragma unroll
    for (int k = 0; k < SHD; ++k) op[k] = o[k];

    float rc  = fminf(r, RMAX);
    float cut = 0.5f * (__cosf(PI_F * rc / RMAX) + 1.0f);
    float pre = 0.6324555320336759f; // sqrt(2/5)
    float* rbp = rb + (size_t)e * NB;
    #pragma unroll
    for (int n = 1; n <= NB; ++n)
        rbp[n-1] = pre * __sinf((float)n * PI_F * r / RMAX) * inv * cut;
}

// ---------------- radial MLP layer 1: h = silu(rb @ W1 + b1), store bf16 ----------------
__global__ __launch_bounds__(256)
void k_radial_mlp1(const float* __restrict__ rb, const float* __restrict__ W1,
                   const float* __restrict__ b1, __bf16* __restrict__ h) {
    int e = blockIdx.x * blockDim.x + threadIdx.x;
    if (e >= EE) return;
    float r[NB];
    #pragma unroll
    for (int k = 0; k < NB; ++k) r[k] = rb[(size_t)e*NB + k];
    v8bf* hp = (v8bf*)(h + (size_t)e * HR);
    #pragma unroll
    for (int c = 0; c < HR/8; ++c) {
        v8bf v;
        #pragma unroll
        for (int jj = 0; jj < 8; ++jj) {
            int j = c*8 + jj;
            float acc = b1[j];
            #pragma unroll
            for (int k = 0; k < NB; ++k) acc = fmaf(r[k], W1[k*HR + j], acc);
            v[jj] = (__bf16)silu_f(acc);
        }
        hp[c] = v;
    }
}

// ---------------- init feats: feats[:,:,0] = emb_W[specie], rest 0; scal = emb ----------------
__global__ __launch_bounds__(256)
void k_init_feats(const float* __restrict__ embW, const int* __restrict__ specie,
                  float* __restrict__ feats, float* __restrict__ scal) {
    int id = blockIdx.x * blockDim.x + threadIdx.x;
    if (id >= NN * FF) return;
    int n = id >> 7, f = id & 127;
    float v = embW[specie[n]*FF + f];
    float* fp = feats + (size_t)n*FF*SHD + (size_t)f*SHD;
    fp[0] = v;
    #pragma unroll
    for (int k = 1; k < SHD; ++k) fp[k] = 0.0f;
    scal[(size_t)n*FF + f] = v;
}

// ---------------- fused: rw = h@W2 (WMMA, 16-edge tile) + message scatter ----------------
// one block = 16 edges; 8 waves; rw tile 16x512 built in LDS, then atomic scatter to agg.
// w2t is transposed: [512][64] bf16, so B fragments are contiguous 32B loads.
__global__ __launch_bounds__(256)
void k_msg_scatter(const __bf16* __restrict__ h,       // [E,64]
                   const __bf16* __restrict__ w2t,     // [512,64] transposed
                   const float* __restrict__ sh,       // [E,16]
                   const float* __restrict__ scal,     // [N,F]
                   const int* __restrict__ senders,
                   const int* __restrict__ receivers,
                   float* __restrict__ agg) {          // [N,F,16]
    __shared__ float rw[16][512];      // 32 KB
    __shared__ float ssh[16][16];
    __shared__ float sscal[16][128];   // 8 KB
    __shared__ int   srecv[16];

    const int e0   = blockIdx.x * 16;
    const int lane = threadIdx.x & 31;
    const int wave = threadIdx.x >> 5;

    // A-fragment mapping (16-bit 16x32): row = lane&15;
    // lanes<16 -> K in {0..15}; lanes>=16 -> K in {16..31} (matches B)
    const int row  = lane & 15;
    const int kg0  = (lane < 16) ? 0 : 16;
    const int kg1  = (lane < 16) ? 8 : 24;
    // B-fragment (32x16): col = lane&15; this code uses lanes<16 rows K0..15, lanes>=16 K16..31
    const int col  = lane & 15;
    const int koff = (lane < 16) ? 0 : 16;
    const int mbase = (lane < 16) ? 0 : 8;

    // A fragments are invariant across N-tiles: hoist (2 K-chunks).
    const __bf16* hr = h + (size_t)(e0 + row) * HR;
    v16bf a0 = make_frag(hr + kg0,      hr + kg1);
    v16bf a1 = make_frag(hr + 32 + kg0, hr + 32 + kg1);

    // each wave computes 4 N-tiles of the 16x512 output
    for (int q = 0; q < 4; ++q) {
        int t = wave * 4 + q;
        const __bf16* wb = w2t + (size_t)(t*16 + col) * HR + koff;
        if (q < 3)
            __builtin_prefetch(w2t + (size_t)((t+1)*16 + col) * HR, 0, 3);
        v16bf b0 = *(const v16bf*)(wb);        // K  0..15 (this lane's half-range)
        v16bf b1 = *(const v16bf*)(wb + 32);   // K 32..47 (+koff)
        v8f c = {};
        c = __builtin_amdgcn_wmma_f32_16x16x32_bf16(false, a0, false, b0,
                                                    (short)0, c, false, false);
        c = __builtin_amdgcn_wmma_f32_16x16x32_bf16(false, a1, false, b1,
                                                    (short)0, c, false, false);
        #pragma unroll
        for (int r = 0; r < 8; ++r)
            rw[mbase + r][t*16 + col] = c[r];
    }

    // stage sh, sender scalars, receivers
    for (int idx = threadIdx.x; idx < 16*16; idx += 256)
        ssh[idx >> 4][idx & 15] = sh[(size_t)(e0 + (idx >> 4))*SHD + (idx & 15)];
    if (threadIdx.x < 16) srecv[threadIdx.x] = receivers[e0 + threadIdx.x];
    for (int idx = threadIdx.x; idx < 16*128; idx += 256) {
        int e = idx >> 7, f = idx & 127;
        sscal[e][f] = scal[(size_t)senders[e0 + e]*FF + f];
    }
    __syncthreads();

    // scatter: 16 edges x 128 f x 16 k atomics (agg is L2-resident: 82 MB < 192 MB)
    #pragma unroll
    for (int p = 0; p < 8; ++p) {
        int idx = threadIdx.x + p*256;     // 0..2047 -> (e,f)
        int e = idx >> 7, f = idx & 127;
        float s = sscal[e][f];
        float rwl[4];
        #pragma unroll
        for (int l = 0; l < 4; ++l) rwl[l] = rw[e][f*4 + l];
        float* dst = agg + (size_t)srecv[e]*FF*SHD + (size_t)f*SHD;
        #pragma unroll
        for (int k = 0; k < 16; ++k) {
            float v = s * rwl[c_lmap[k]] * ssh[e][k];
            __hip_atomic_fetch_add(dst + k, v, __ATOMIC_RELAXED, __HIP_MEMORY_SCOPE_AGENT);
        }
    }
}

// ---------------- per-node species linear via WMMA ----------------
// out[n,g,off_l+m] = scale * sum_f in[n,f,off_l+m] * W[s,l,f,g]
// computed as out^T(m x 128) = A(m x 128, pad 16) @ W_l(128 x 128); one block per node.
// Wt is transposed per block: [g][f] bf16, so B fragments are contiguous 32B loads.
__global__ __launch_bounds__(256)
void k_species_linear(const float* __restrict__ in,    // [N,F,16]
                      const __bf16* __restrict__ Wt,   // [NSPEC,4,F(g),F(f)] (layer slice)
                      const int* __restrict__ specie,
                      float* __restrict__ outp,        // [N,F,16]
                      float scale) {
    __shared__ __attribute__((aligned(32))) __bf16 A[4][16][128];   // 16 KB
    const int n = blockIdx.x;
    const int s = specie[n];

    // fill A tiles (vectorized: 8 bf16 per store)
    for (int cidx = threadIdx.x; cidx < 4*16*16; cidx += 256) {
        int l = cidx >> 8, m = (cidx >> 4) & 15, fc = cidx & 15;
        int f0 = fc * 8;
        v8bf v;
        if (m < c_ml[l]) {
            const float* ip = in + (size_t)n*FF*SHD + (size_t)f0*SHD + c_off[l] + m;
            #pragma unroll
            for (int j = 0; j < 8; ++j) v[j] = (__bf16)ip[(size_t)j*SHD];
        } else {
            #pragma unroll
            for (int j = 0; j < 8; ++j) v[j] = (__bf16)0.0f;
        }
        *(v8bf*)&A[l][m][f0] = v;
    }
    __syncthreads();

    const int lane = threadIdx.x & 31;
    const int wave = threadIdx.x >> 5;
    const int row  = lane & 15;
    const int kg0  = (lane < 16) ? 0 : 16;
    const int kg1  = (lane < 16) ? 8 : 24;
    const int col  = lane & 15;
    const int koff = (lane < 16) ? 0 : 16;
    const int mbase = (lane < 16) ? 0 : 8;

    // wave w handles N-tile t=w for all 4 l-blocks
    const int t = wave;
    for (int l = 0; l < 4; ++l) {
        const __bf16* Wl = Wt + ((size_t)s*4 + l) * FF * FF;
        const __bf16* wb = Wl + (size_t)(t*16 + col) * FF + koff;
        if (l < 3)
            __builtin_prefetch(Wt + ((size_t)s*4 + l + 1) * FF * FF
                                  + (size_t)(t*16 + col) * FF, 0, 3);
        v8f c = {};
        #pragma unroll
        for (int kc = 0; kc < 4; ++kc) {
            int kb = kc * 32;
            v16bf a = make_frag(&A[l][row][kb + kg0], &A[l][row][kb + kg1]);
            v16bf b = *(const v16bf*)(wb + kb);    // K kb+koff .. kb+koff+15, contiguous
            c = __builtin_amdgcn_wmma_f32_16x16x32_bf16(false, a, false, b,
                                                        (short)0, c, false, false);
        }
        int off = c_off[l], ml = c_ml[l];
        #pragma unroll
        for (int r = 0; r < 8; ++r) {
            int m = mbase + r;
            if (m < ml)
                outp[(size_t)n*FF*SHD + (size_t)(t*16 + col)*SHD + off + m] = c[r] * scale;
        }
    }
}

// ---------------- poly gating + feature update ----------------
__global__ __launch_bounds__(256)
void k_update(const float* __restrict__ newf, const float* __restrict__ scf,
              const float* __restrict__ prodc,   // [NSPEC,3,F] (layer slice)
              const int* __restrict__ specie,
              float* __restrict__ feats, float* __restrict__ scal, int use_sc) {
    int id = blockIdx.x * blockDim.x + threadIdx.x;
    if (id >= NN * FF) return;
    int n = id >> 7, f = id & 127;
    int s = specie[n];
    const float* cp = prodc + (size_t)s*3*FF;
    const float* np = newf + (size_t)n*FF*SHD + (size_t)f*SHD;
    float sv = np[0];
    float poly = cp[f] + cp[FF + f]*sv + cp[2*FF + f]*sv*sv;
    float* fp = feats + (size_t)n*FF*SHD + (size_t)f*SHD;
    const float* sp = scf + (size_t)n*FF*SHD + (size_t)f*SHD;
    #pragma unroll
    for (int k = 0; k < SHD; ++k) {
        float v = np[k] * poly;
        if (use_sc) v += sp[k];
        fp[k] = v;
    }
    scal[(size_t)n*FF + f] = fp[0];
}

// ---------------- readouts ----------------
__global__ __launch_bounds__(256)
void k_readout0(const float* __restrict__ scal, const float* __restrict__ w, float* __restrict__ out) {
    int n = blockIdx.x * blockDim.x + threadIdx.x;
    if (n >= NN) return;
    float acc = 0.0f;
    const float* sp = scal + (size_t)n*FF;
    #pragma unroll 8
    for (int f = 0; f < FF; ++f) acc = fmaf(sp[f], w[f], acc);
    out[n*NLAY + 0] = acc;
}

__global__ __launch_bounds__(256)
void k_readout1(const float* __restrict__ scal, const float* __restrict__ w1,
                const float* __restrict__ w2, float* __restrict__ out) {
    int n = blockIdx.x * blockDim.x + threadIdx.x;
    if (n >= NN) return;
    float acc[16];
    #pragma unroll
    for (int j = 0; j < 16; ++j) acc[j] = 0.0f;
    const float* sp = scal + (size_t)n*FF;
    for (int f = 0; f < FF; ++f) {
        float sv = sp[f];
        #pragma unroll
        for (int j = 0; j < 16; ++j) acc[j] = fmaf(sv, w1[f*16 + j], acc[j]);
    }
    float r = 0.0f;
    #pragma unroll
    for (int j = 0; j < 16; ++j) r = fmaf(silu_f(acc[j]), w2[j], r);
    out[n*NLAY + 1] = r;
}

// ---------------- host orchestration ----------------
extern "C" void kernel_launch(void* const* d_in, const int* in_sizes, int n_in,
                              void* d_out, int out_size, void* d_ws, size_t ws_size,
                              hipStream_t stream) {
    const float* vectors   = (const float*)d_in[0];
    const int*   specie    = (const int*)  d_in[1];
    const int*   senders   = (const int*)  d_in[2];
    const int*   receivers = (const int*)  d_in[3];
    const float* embW      = (const float*)d_in[4];
    const float* rW1       = (const float*)d_in[5];   // [NL,8,64]
    const float* rb1       = (const float*)d_in[6];   // [NL,64]
    const float* rW2       = (const float*)d_in[7];   // [NL,64,512]
    const float* linW      = (const float*)d_in[8];   // [NL,10,4,128,128]
    const float* prodc     = (const float*)d_in[9];   // [NL,10,3,128]
    const float* scW       = (const float*)d_in[10];  // [NL,10,4,128,128]
    const float* ro0W      = (const float*)d_in[11];
    const float* ro1W1     = (const float*)d_in[12];
    const float* ro1W2     = (const float*)d_in[13];
    float* out = (float*)d_out;

    char* ws = (char*)d_ws;
    size_t off = 0;
    auto alloc = [&](size_t bytes) -> void* {
        void* p = ws + off;
        off += (bytes + 255) & ~(size_t)255;
        return p;
    };
    float*  sh    = (float*) alloc((size_t)EE*SHD*4);
    float*  rb    = (float*) alloc((size_t)EE*NB*4);
    __bf16* hbf   = (__bf16*)alloc((size_t)EE*HR*2);
    float*  feats = (float*) alloc((size_t)NN*FF*SHD*4);
    float*  scal  = (float*) alloc((size_t)NN*FF*4);
    float*  agg   = (float*) alloc((size_t)NN*FF*SHD*4);
    float*  newf  = (float*) alloc((size_t)NN*FF*SHD*4);
    float*  scf   = (float*) alloc((size_t)NN*FF*SHD*4);
    __bf16* w2t   = (__bf16*)alloc((size_t)NLAY*512*HR*2);
    __bf16* lint  = (__bf16*)alloc((size_t)NLAY*NSPEC*4*FF*FF*2);
    __bf16* sct   = (__bf16*)alloc((size_t)NLAY*NSPEC*4*FF*FF*2);
    (void)ws_size; (void)in_sizes; (void)n_in; (void)out_size;

    k_cvt_w2t<<<512, 256, 0, stream>>>(rW2, w2t);
    k_cvt_lint<<<2048, 256, 0, stream>>>(linW, lint);
    k_cvt_lint<<<2048, 256, 0, stream>>>(scW,  sct);

    k_edge_prep<<<(EE + 255)/256, 256, 0, stream>>>(vectors, sh, rb);
    k_init_feats<<<(NN*FF + 255)/256, 256, 0, stream>>>(embW, specie, feats, scal);

    for (int i = 0; i < NLAY; ++i) {
        k_zero_f32<<<4096, 256, 0, stream>>>(agg, (long)NN*FF*SHD);
        k_radial_mlp1<<<(EE + 255)/256, 256, 0, stream>>>(rb, rW1 + (size_t)i*NB*HR,
                                                          rb1 + (size_t)i*HR, hbf);
        k_msg_scatter<<<EE/16, 256, 0, stream>>>(hbf, w2t + (size_t)i*512*HR, sh, scal,
                                                 senders, receivers, agg);
        k_species_linear<<<NN, 256, 0, stream>>>(agg, lint + (size_t)i*NSPEC*4*FF*FF,
                                                 specie, newf, EPS_C);
        if (i > 0)
            k_species_linear<<<NN, 256, 0, stream>>>(feats, sct + (size_t)i*NSPEC*4*FF*FF,
                                                     specie, scf, 1.0f);
        k_update<<<(NN*FF + 255)/256, 256, 0, stream>>>(newf, scf, prodc + (size_t)i*NSPEC*3*FF,
                                                        specie, feats, scal, (i > 0) ? 1 : 0);
        if (i == 0)
            k_readout0<<<(NN + 255)/256, 256, 0, stream>>>(scal, ro0W, out);
        else
            k_readout1<<<(NN + 255)/256, 256, 0, stream>>>(scal, ro1W1, ro1W2, out);
    }
}
